// Attention_34608846471556
// MI455X (gfx1250) — matrix operand order repaired and checked by reading, not compiled, not run on hardware
//
#include <hip/hip_runtime.h>
#include <hip/hip_bf16.h>

typedef __bf16 bf16;
typedef __attribute__((ext_vector_type(16))) __bf16 bf16x16;
typedef __attribute__((ext_vector_type(8)))  __bf16 bf16x8;
typedef __attribute__((ext_vector_type(8)))  float  f32x8;

#define BDIM   2
#define NSEQ   2048
#define DMODEL 1024
#define HEADS  16
#define DHEAD  64
#define INNER  1024
#define NQKV   3072
#define MROWS  (BDIM * NSEQ)   /* 4096 */
#define SCALE  0.125f          /* 64^-0.5 */

static __device__ __forceinline__ int lane_id() { return (int)(threadIdx.x & 31u); }

static __device__ __forceinline__ f32x8 wmma_bf16(bf16x16 a, bf16x16 b, f32x8 c) {
  // D = A(16x32) * B(32x16) + C, f32 accumulate
  return __builtin_amdgcn_wmma_f32_16x16x32_bf16(false, a, false, b, (short)0, c, false, false);
}

// A fragment: 16x32 bf16 tile, row-major at p0, leading dim ld (elements).
// Per ISA: lane m=lane&15; kb=8*(lane>>4); elems 0..7 -> K=kb.., elems 8..15 -> K=16+kb..
static __device__ __forceinline__ bf16x16 load_a_frag(const bf16* p0, int ld) {
  const int lane = lane_id();
  const bf16* p = p0 + (size_t)(lane & 15) * ld + ((lane >> 4) << 3);
  bf16x8 lo = *(const bf16x8*)(p);
  bf16x8 hi = *(const bf16x8*)(p + 16);
  bf16x16 r;
#pragma unroll
  for (int i = 0; i < 8; ++i) { r[i] = lo[i]; r[i + 8] = hi[i]; }
  return r;
}

// B fragment sourced from Bt[n][k] (B transposed, row-major over k, ld elements).
// Per ISA: lane n=lane&15; kb=8*(lane>>4); elems 0..7 -> K=kb.., elems 8..15 -> K=16+kb..
static __device__ __forceinline__ bf16x16 load_bt_frag(const bf16* p0, int ld) {
  const int lane = lane_id();
  const bf16* p = p0 + (size_t)(lane & 15) * ld + ((lane >> 4) << 3);
  bf16x8 lo = *(const bf16x8*)(p);
  bf16x8 hi = *(const bf16x8*)(p + 16);
  bf16x16 r;
#pragma unroll
  for (int i = 0; i < 8; ++i) { r[i] = lo[i]; r[i + 8] = hi[i]; }
  return r;
}

static __device__ __forceinline__ float half_max16(float v) {
  v = fmaxf(v, __shfl_xor(v, 1, 32));
  v = fmaxf(v, __shfl_xor(v, 2, 32));
  v = fmaxf(v, __shfl_xor(v, 4, 32));
  v = fmaxf(v, __shfl_xor(v, 8, 32));
  return v;
}
static __device__ __forceinline__ float half_sum16(float v) {
  v += __shfl_xor(v, 1, 32);
  v += __shfl_xor(v, 2, 32);
  v += __shfl_xor(v, 4, 32);
  v += __shfl_xor(v, 8, 32);
  return v;
}

// ---------------- prep kernels ----------------
__global__ void k_cast_bf16(const float* __restrict__ in, bf16* __restrict__ out, int n) {
  int i = blockIdx.x * blockDim.x + threadIdx.x;
  if (i < n) out[i] = (bf16)in[i];
}

// in[rows][cols] f32 -> out[cols][rows] bf16
__global__ void k_transpose_bf16(const float* __restrict__ in, bf16* __restrict__ out,
                                 int rows, int cols) {
  int i = blockIdx.x * blockDim.x + threadIdx.x;
  if (i < rows * cols) {
    int c = i / rows, r = i - c * rows;
    out[(size_t)c * rows + r] = (bf16)in[(size_t)r * cols + c];
  }
}

// ---------------- QKV projection GEMM ----------------
// C[4096 x 3072] = Xb[4096 x 1024] * WqT^T ; scatter into Q/K (B,H,N,D) and Vt (B,H,D,N).
__global__ __launch_bounds__(128) void k_qkv_gemm(
    const bf16* __restrict__ Xb, const bf16* __restrict__ WqT,
    bf16* __restrict__ Q, bf16* __restrict__ Km, bf16* __restrict__ Vt) {
  const int wave = threadIdx.x >> 5;
  const int lane = lane_id();
  const int hf = lane >> 4, nl = lane & 15;
  const int m0 = blockIdx.x * 16;
  const int n0 = (blockIdx.y * 4 + wave) * 64;

  const bf16* Ap = Xb + (size_t)m0 * DMODEL;
  const bf16* Bp = WqT + (size_t)n0 * DMODEL;

  f32x8 zero = {};
  f32x8 acc[4];
#pragma unroll
  for (int g = 0; g < 4; ++g) acc[g] = zero;

  for (int kk = 0; kk < DMODEL; kk += 32) {
    bf16x16 a = load_a_frag(Ap + kk, DMODEL);
#pragma unroll
    for (int g = 0; g < 4; ++g) {
      bf16x16 bfr = load_bt_frag(Bp + (size_t)(g * 16) * DMODEL + kk, DMODEL);
      acc[g] = wmma_bf16(a, bfr, acc[g]);
    }
  }

#pragma unroll
  for (int g = 0; g < 4; ++g) {
#pragma unroll
    for (int r = 0; r < 8; ++r) {
      int row = m0 + r + 8 * hf;          // [0,4096)
      int col = n0 + g * 16 + nl;         // [0,3072)
      float v = acc[g][r];
      int bb = row >> 11, nn = row & (NSEQ - 1);
      int which = col >> 10, cc = col & 1023;
      int hh = cc >> 6, dd = cc & 63;
      size_t bhh = (size_t)(bb * HEADS + hh);
      if (which == 0)      Q [(bhh * NSEQ + nn) * DHEAD + dd] = (bf16)(v * SCALE);
      else if (which == 1) Km[(bhh * NSEQ + nn) * DHEAD + dd] = (bf16)v;
      else                 Vt[(bhh * DHEAD + dd) * NSEQ + nn] = (bf16)v;
    }
  }
}

// ---------------- flash attention (per-wave 16 query rows) ----------------
__global__ __launch_bounds__(128) void k_attention(
    const bf16* __restrict__ Q, const bf16* __restrict__ Km,
    const bf16* __restrict__ Vt, bf16* __restrict__ AO) {
  __shared__ bf16 Plds[4][16 * 32];
  const int wave = threadIdx.x >> 5;
  const int lane = lane_id();
  const int hf = lane >> 4, nl = lane & 15;
  const int bh = blockIdx.x;               // b*HEADS + h
  const int b = bh >> 4, h = bh & 15;
  const int qbase = (blockIdx.y * 4 + wave) * 16;

  const bf16* Qp = Q + ((size_t)bh * NSEQ + qbase) * DHEAD;
  const bf16* Kp = Km + (size_t)bh * NSEQ * DHEAD;
  const bf16* Vp = Vt + (size_t)bh * DHEAD * NSEQ;

  bf16x16 aq0 = load_a_frag(Qp, DHEAD);        // d = 0..31
  bf16x16 aq1 = load_a_frag(Qp + 32, DHEAD);   // d = 32..63

  f32x8 o0 = {}, o1 = {}, o2 = {}, o3 = {};
  f32x8 mrow, lrow;
#pragma unroll
  for (int r = 0; r < 8; ++r) { mrow[r] = -3.0e38f; lrow[r] = 0.0f; }

  bf16* Pw = &Plds[wave][0];

  for (int j = 0; j < NSEQ; j += 32) {
    f32x8 c0 = {}, c1 = {};
    // S = Q * K^T  (K-dim = d = 64 -> two WMMA accumulations; keys split 16+16)
    c0 = wmma_bf16(aq0, load_bt_frag(Kp + (size_t)j * DHEAD, DHEAD), c0);
    c0 = wmma_bf16(aq1, load_bt_frag(Kp + (size_t)j * DHEAD + 32, DHEAD), c0);
    c1 = wmma_bf16(aq0, load_bt_frag(Kp + (size_t)(j + 16) * DHEAD, DHEAD), c1);
    c1 = wmma_bf16(aq1, load_bt_frag(Kp + (size_t)(j + 16) * DHEAD + 32, DHEAD), c1);

    // online softmax: row stats live per C-fragment register r (row = r + 8*hf)
    f32x8 p0, p1, alpha;
#pragma unroll
    for (int r = 0; r < 8; ++r) {
      float mn = half_max16(fmaxf(c0[r], c1[r]));
      float mt = fmaxf(mrow[r], mn);
      alpha[r] = __expf(mrow[r] - mt);
      p0[r] = __expf(c0[r] - mt);
      p1[r] = __expf(c1[r] - mt);
      lrow[r] = lrow[r] * alpha[r] + half_sum16(p0[r] + p1[r]);
      mrow[r] = mt;
    }
    o0 *= alpha; o1 *= alpha; o2 *= alpha; o3 *= alpha;

    // C-fragment -> A-fragment relayout of P via LDS (16x32 bf16, row-major)
#pragma unroll
    for (int r = 0; r < 8; ++r) {
      Pw[(r + 8 * hf) * 32 + nl]      = (bf16)p0[r];
      Pw[(r + 8 * hf) * 32 + 16 + nl] = (bf16)p1[r];
    }
    __asm__ volatile("s_wait_dscnt 0" ::: "memory");
    bf16x16 ap = load_a_frag(Pw, 32);

    // O += P * V   (K-dim = 32 keys; N-dim = d, 4 x 16-wide groups)
    o0 = wmma_bf16(ap, load_bt_frag(Vp + (size_t)(0 * 16) * NSEQ + j, NSEQ), o0);
    o1 = wmma_bf16(ap, load_bt_frag(Vp + (size_t)(1 * 16) * NSEQ + j, NSEQ), o1);
    o2 = wmma_bf16(ap, load_bt_frag(Vp + (size_t)(2 * 16) * NSEQ + j, NSEQ), o2);
    o3 = wmma_bf16(ap, load_bt_frag(Vp + (size_t)(3 * 16) * NSEQ + j, NSEQ), o3);
  }

  f32x8 inv;
#pragma unroll
  for (int r = 0; r < 8; ++r) inv[r] = 1.0f / lrow[r];
  o0 *= inv; o1 *= inv; o2 *= inv; o3 *= inv;

  bf16* AOp = AO + ((size_t)(b * NSEQ + qbase)) * INNER + h * DHEAD;
#pragma unroll
  for (int r = 0; r < 8; ++r) {
    int row = r + 8 * hf;
    AOp[(size_t)row * INNER + 0 * 16 + nl] = (bf16)o0[r];
    AOp[(size_t)row * INNER + 1 * 16 + nl] = (bf16)o1[r];
    AOp[(size_t)row * INNER + 2 * 16 + nl] = (bf16)o2[r];
    AOp[(size_t)row * INNER + 3 * 16 + nl] = (bf16)o3[r];
  }
}

// ---------------- output projection GEMM + bias ----------------
__global__ __launch_bounds__(128) void k_out_gemm(
    const bf16* __restrict__ AO, const bf16* __restrict__ WoT,
    const float* __restrict__ bias, float* __restrict__ Out) {
  const int wave = threadIdx.x >> 5;
  const int lane = lane_id();
  const int hf = lane >> 4, nl = lane & 15;
  const int m0 = blockIdx.x * 16;
  const int n0 = (blockIdx.y * 4 + wave) * 64;

  const bf16* Ap = AO + (size_t)m0 * INNER;
  const bf16* Bp = WoT + (size_t)n0 * INNER;

  f32x8 zero = {};
  f32x8 acc[4];
#pragma unroll
  for (int g = 0; g < 4; ++g) acc[g] = zero;

  for (int kk = 0; kk < INNER; kk += 32) {
    bf16x16 a = load_a_frag(Ap + kk, INNER);
#pragma unroll
    for (int g = 0; g < 4; ++g) {
      bf16x16 bfr = load_bt_frag(Bp + (size_t)(g * 16) * INNER + kk, INNER);
      acc[g] = wmma_bf16(a, bfr, acc[g]);
    }
  }

#pragma unroll
  for (int g = 0; g < 4; ++g) {
#pragma unroll
    for (int r = 0; r < 8; ++r) {
      int row = m0 + r + 8 * hf;
      int col = n0 + g * 16 + nl;
      Out[(size_t)row * DMODEL + col] = acc[g][r] + bias[col];
    }
  }
}

extern "C" void kernel_launch(void* const* d_in, const int* in_sizes, int n_in,
                              void* d_out, int out_size, void* d_ws, size_t ws_size,
                              hipStream_t stream) {
  (void)in_sizes; (void)n_in; (void)out_size; (void)ws_size;
  const float* x     = (const float*)d_in[0];
  const float* w_qkv = (const float*)d_in[1];
  const float* w_out = (const float*)d_in[2];
  const float* b_out = (const float*)d_in[3];
  float* out = (float*)d_out;

  // workspace layout (bf16 buffers, 1 MB-aligned offsets; ~48 MB total)
  const size_t MB = 1024 * 1024;
  char* ws = (char*)d_ws;
  bf16* xb  = (bf16*)(ws + 0 * MB);   // 4096 x 1024
  bf16* wqT = (bf16*)(ws + 8 * MB);   // 3072 x 1024  (W_qkv^T)
  bf16* woT = (bf16*)(ws + 14 * MB);  // 1024 x 1024  (W_out^T)
  bf16* Q   = (bf16*)(ws + 16 * MB);  // [B,H,N,D], pre-scaled
  bf16* Km  = (bf16*)(ws + 24 * MB);  // [B,H,N,D]
  bf16* Vt  = (bf16*)(ws + 32 * MB);  // [B,H,D,N]
  bf16* AO  = (bf16*)(ws + 40 * MB);  // [B,N,INNER]

  const int nx = MROWS * DMODEL;
  k_cast_bf16<<<(nx + 255) / 256, 256, 0, stream>>>(x, xb, nx);
  k_transpose_bf16<<<(DMODEL * NQKV + 255) / 256, 256, 0, stream>>>(w_qkv, wqT, DMODEL, NQKV);
  k_transpose_bf16<<<(DMODEL * DMODEL + 255) / 256, 256, 0, stream>>>(w_out, woT, DMODEL, DMODEL);

  k_qkv_gemm<<<dim3(MROWS / 16, NQKV / 256), 128, 0, stream>>>(xb, wqT, Q, Km, Vt);
  k_attention<<<dim3(BDIM * HEADS, NSEQ / 64), 128, 0, stream>>>(Q, Km, Vt, AO);
  k_out_gemm<<<dim3(MROWS / 16, DMODEL / 256), 128, 0, stream>>>(AO, woT, b_out, out);
}
